// BiGRU_47725676593567
// MI455X (gfx1250) — matrix operand order repaired and checked by reading, not compiled, not run on hardware
//
#include <hip/hip_runtime.h>
#include <hip/hip_bf16.h>

typedef __attribute__((ext_vector_type(16))) _Float16 v16h;
typedef __attribute__((ext_vector_type(8)))  _Float16 v8h;
typedef __attribute__((ext_vector_type(8)))  float    v8f;

#define B_   32
#define S_   512
#define I_   512
#define H_   1024
#define G3H  3072   // 3*H
#define H2   2048   // 2*H
#define MTOT (B_*S_)

#define STR2(x) #x
#define STR(x) STR2(x)
#if defined(__AMDGCN__)
#pragma message("probe: device clang major = " STR(__clang_major__))
#endif

// ---------------- async global->LDS copy (16B per lane) ----------------
// Signature learned from round-2 diagnostic: param0 is v4i in AS1 (global src),
// param1 v4i in AS3 (LDS dst), then imm offset + cpol.
#if __has_builtin(__builtin_amdgcn_global_load_async_to_lds_b128) && \
    __has_builtin(__builtin_amdgcn_s_wait_asynccnt)
#define USE_ASYNC_LDS 1
typedef int async_v4i __attribute__((vector_size(16)));
typedef __attribute__((address_space(1))) async_v4i* async_gp_t;
typedef __attribute__((address_space(3))) async_v4i* async_lp_t;
#endif

__device__ __forceinline__ void async_cp16(void* lds_dst, const void* gsrc) {
#ifdef USE_ASYNC_LDS
  __builtin_amdgcn_global_load_async_to_lds_b128((async_gp_t)gsrc,
                                                 (async_lp_t)lds_dst, 0, 0);
#else
  *(v8h*)lds_dst = *(const v8h*)gsrc;
#endif
}
__device__ __forceinline__ void async_wait_all() {
#ifdef USE_ASYNC_LDS
  __builtin_amdgcn_s_wait_asynccnt(0);
#endif
}

// ---------------- Tensor Data Mover (TDM) for the h-state staging ----------------
#if __has_builtin(__builtin_amdgcn_tensor_load_to_lds) && \
    __has_builtin(__builtin_amdgcn_s_wait_tensorcnt)
#define USE_TDM 1
typedef unsigned tdm_u32x4 __attribute__((ext_vector_type(4)));
typedef int      tdm_i32x4 __attribute__((ext_vector_type(4)));
typedef int      tdm_i32x8 __attribute__((ext_vector_type(8)));
#endif

// h row layout in LDS after TDM padding: 8B pad after every 1024B ->
// element k of a row lives at (k + (k>=512 ? 4 : 0)), row stride 1032 halves.
__device__ __forceinline__ int hpad(int k) { return k + ((k >= 512) ? 4 : 0); }

#ifdef USE_TDM
// Move 32 x 2048B (h_prev, f16) global -> LDS with 8B pad per 1024B.
// data_size=8B: tile 256 x 32, contiguous (stride 256).
__device__ __forceinline__ void tdm_load_h(void* lds_dst, const void* gsrc) {
  tdm_u32x4 g0;
  tdm_i32x8 g1;
  tdm_i32x4 gz = {0, 0, 0, 0};
  unsigned long long ga = (unsigned long long)(uintptr_t)gsrc;
  g0[0] = 1u;                                   // count=1, user descriptor
  g0[1] = (unsigned)(uintptr_t)lds_dst;         // lds_addr (flat LDS addr truncates to offset)
  g0[2] = (unsigned)ga;                         // global_addr[31:0]
  g0[3] = (unsigned)((ga >> 32) & 0x01FFFFFFu) | (2u << 30);  // addr[56:32] | type=2
  g1[0] = (int)((3u << 16)       // data_size = 8B
              | (1u << 20)       // pad_enable
              | (7u << 22)       // pad_interval: 256 DWORDs = 1024B
              | (1u << 25));     // pad_amount: 2 DWORDs = 8B
  g1[1] = (int)(256u << 16);     // tensor_dim0 = 256 (low16 at bit48)
  g1[2] = (int)(32u << 16);      // tensor_dim0 hi=0 | tensor_dim1 = 32
  g1[3] = (int)(256u << 16);     // tensor_dim1 hi=0 | tile_dim0 = 256
  g1[4] = 32;                    // tile_dim1 = 32, tile_dim2 = 0
  g1[5] = 256;                   // tensor_dim0_stride lo32 = 256
  g1[6] = 0;                     // stride hi | tensor_dim1_stride lo
  g1[7] = 0;
#if __clang_major__ >= 23
  tdm_i32x8 gz8 = {0, 0, 0, 0, 0, 0, 0, 0};
  __builtin_amdgcn_tensor_load_to_lds(g0, g1, gz, gz, gz8, 0);
#else
  __builtin_amdgcn_tensor_load_to_lds(g0, g1, gz, gz, 0);
#endif
}
#endif

// ---------------- f32 -> f16 elementwise convert ----------------
__global__ void cvt_f32_f16(const float* __restrict__ src, _Float16* __restrict__ dst, long n) {
  long i = (long)blockIdx.x * blockDim.x + threadIdx.x;
  long stride = (long)gridDim.x * blockDim.x;
  for (; i < n; i += stride) dst[i] = (_Float16)src[i];
}

// ---------------- input-projection GEMM: gx = X(MxK,f16) * W(NxK,f16)^T + bias ----------------
#define BM 128
#define BN 128
#define BK 32
#define LDT 40   // padded LDS row stride in halves

__global__ __launch_bounds__(256)
void gemm_gx(const _Float16* __restrict__ X,
             const _Float16* __restrict__ Wf, const _Float16* __restrict__ Wb,
             const float* __restrict__ bf,    const float* __restrict__ bb,
             float* __restrict__ outf,        float* __restrict__ outb,
             int K) {
  const _Float16* W   = (blockIdx.z == 0) ? Wf : Wb;
  const float*   bias = (blockIdx.z == 0) ? bf : bb;
  float*         out  = (blockIdx.z == 0) ? outf : outb;

  __shared__ _Float16 At[BM][LDT];
  __shared__ _Float16 Bt[BN][LDT];

  const int tid  = threadIdx.x;
  const int lane = tid & 31;
  const int w    = tid >> 5;   // 0..7
  const int wm   = w & 3;      // 32-row M subtile
  const int wn   = w >> 2;     // 64-col N half

  const int m0 = blockIdx.y * BM;
  const int n0 = blockIdx.x * BN;

  v8f acc[2][4] = {};

  for (int k0 = 0; k0 < K; k0 += BK) {
    #pragma unroll
    for (int rr = 0; rr < 2; ++rr) {
      int idx = tid + rr * 256;        // 0..511
      int row = idx >> 2;              // 0..127
      int c8  = (idx & 3) << 3;        // 0,8,16,24
      async_cp16(&At[row][c8], &X[(long)(m0 + row) * K + k0 + c8]);
      async_cp16(&Bt[row][c8], &W[(long)(n0 + row) * K + k0 + c8]);
    }
    async_wait_all();
    __syncthreads();

    const int am = lane & 15;
    const int kh = (lane >> 4) << 3;
    const int kb = (lane >> 4) << 4;

    v16h a[2];
    #pragma unroll
    for (int mi = 0; mi < 2; ++mi) {
      v8h alo = *(const v8h*)&At[wm * 32 + mi * 16 + am][kh];
      v8h ahi = *(const v8h*)&At[wm * 32 + mi * 16 + am][kh + 16];
      a[mi] = __builtin_shufflevector(alo, ahi, 0,1,2,3,4,5,6,7,8,9,10,11,12,13,14,15);
    }
    #pragma unroll
    for (int t = 0; t < 4; ++t) {
      v8h blo = *(const v8h*)&Bt[wn * 64 + t * 16 + am][kh];
      v8h bhi = *(const v8h*)&Bt[wn * 64 + t * 16 + am][kh + 16];
      v16h b = __builtin_shufflevector(blo, bhi, 0,1,2,3,4,5,6,7,8,9,10,11,12,13,14,15);
      acc[0][t] = __builtin_amdgcn_wmma_f32_16x16x32_f16(false, a[0], false, b,
                                                         (short)0, acc[0][t], false, false);
      acc[1][t] = __builtin_amdgcn_wmma_f32_16x16x32_f16(false, a[1], false, b,
                                                         (short)0, acc[1][t], false, false);
    }
    __syncthreads();
  }

  #pragma unroll
  for (int mi = 0; mi < 2; ++mi) {
    #pragma unroll
    for (int t = 0; t < 4; ++t) {
      int n = n0 + wn * 64 + t * 16 + (lane & 15);
      float bv = bias[n];
      int mbase = m0 + wm * 32 + mi * 16 + ((lane >> 4) << 3);
      #pragma unroll
      for (int r = 0; r < 8; ++r)
        out[(long)(mbase + r) * G3H + n] = acc[mi][t][r] + bv;
    }
  }
}

// ---------------- persistent grid-synced recurrence ----------------
#define NWG  64
#define RT   192     // 6 waves: (mi in {0,1}) x (gate in {0,1,2})
#define LDH  1032    // padded row stride (halves)

__global__ __launch_bounds__(RT)
void bigru_recur(const float* __restrict__ gxF, const float* __restrict__ gxB,
                 const _Float16* __restrict__ whhF, const _Float16* __restrict__ whhB,
                 const float* __restrict__ bhhF,   const float* __restrict__ bhhB,
                 _Float16* __restrict__ seq,   // (B,S,2H) f16 or null
                 float* __restrict__ outp,     // (B,2H) f32 or null
                 _Float16* __restrict__ hbase, // [2 dir][2 buf][B*H] f16
                 unsigned* __restrict__ counters) {
  extern __shared__ __align__(16) char smem[];
  _Float16* wsl = (_Float16*)smem;                           // [48][LDH]
  _Float16* hb  = (_Float16*)(smem + 48 * LDH * 2);          // [32][LDH], padded 8B/1024B
  float*    ghb = (float*)(smem + (48 + 32) * LDH * 2);      // [3][32][16]

  const int dir  = blockIdx.x / NWG;
  const int wg   = blockIdx.x % NWG;
  const int j0   = wg * 16;
  const int tid  = threadIdx.x;
  const int lane = tid & 31;
  const int w    = tid >> 5;
  const int g    = w % 3;
  const int mi   = w / 3;

  const float*    gx  = dir ? gxB  : gxF;
  const _Float16* whh = dir ? whhB : whhF;
  const float*    bhh = dir ? bhhB : bhhF;
  unsigned* ctr = counters + dir * 16;
  _Float16* h0  = hbase + (long)dir * 2 * B_ * H_;

  // one-time weight slice load: wsl row (g*16+jj) <- whh[g*H + j0+jj][0..1023]
  for (int idx = tid; idx < 48 * (H_ / 8); idx += RT) {
    int row = idx / (H_ / 8);
    int c8  = (idx % (H_ / 8)) * 8;
    int gg = row >> 4, jj = row & 15;
    async_cp16(&wsl[row * LDH + c8],
               &whh[(long)(gg * H_ + j0 + jj) * H_ + c8]);
  }
  async_wait_all();

  const int am = lane & 15;
  const int kh = (lane >> 4) << 3;
  const int kb = (lane >> 4) << 4;
  const _Float16* brow = &wsl[(g * 16 + am) * LDH];

  for (int step = 0; step < S_; ++step) {
    const int s = dir ? (S_ - 1 - step) : step;
    const _Float16* hcur = h0 + (step & 1) * (B_ * H_);
    _Float16*       hnxt = h0 + ((step + 1) & 1) * (B_ * H_);

    // stage h_prev (32x1024 f16) into LDS: one TDM descriptor (DMA engine
    // inserts the 8B/1024B padding), fallback = per-lane async b128 copies.
#ifdef USE_TDM
    if (w == 0) {
      tdm_load_h(hb, hcur);
      __builtin_amdgcn_s_wait_tensorcnt(0);
    }
#else
    for (int idx = tid; idx < 32 * (H_ / 8); idx += RT) {
      int row = idx >> 7;
      int c8  = (idx & 127) << 3;
      async_cp16(&hb[row * LDH + hpad(c8)], &hcur[row * H_ + c8]);
    }
#endif
    // prefetch this step's gx operands; lines arrive while WMMAs run
    for (int e = tid; e < 32 * 16; e += RT) {
      const float* p = gx + ((long)(e >> 4) * S_ + s) * (long)G3H + j0 + (e & 15);
      __builtin_prefetch(p, 0, 1);
      __builtin_prefetch(p + H_, 0, 1);
      __builtin_prefetch(p + 2 * H_, 0, 1);
    }
    async_wait_all();
    __syncthreads();

    // gh tile for (rows mi*16.., gate g, cols j0..j0+15); two accs to pipeline
    const _Float16* arow = &hb[(mi * 16 + am) * LDH];
    v8f acc0 = {}, acc1 = {};
    #pragma unroll 4
    for (int kk = 0; kk < H_; kk += 64) {
      {
        int ko = hpad(kk);
        v8h alo = *(const v8h*)&arow[ko + kh];
        v8h ahi = *(const v8h*)&arow[ko + kh + 16];
        v16h a = __builtin_shufflevector(alo, ahi, 0,1,2,3,4,5,6,7,8,9,10,11,12,13,14,15);
        v8h blo = *(const v8h*)&brow[kk + kh];
        v8h bhi = *(const v8h*)&brow[kk + kh + 16];
        v16h b = __builtin_shufflevector(blo, bhi, 0,1,2,3,4,5,6,7,8,9,10,11,12,13,14,15);
        acc0 = __builtin_amdgcn_wmma_f32_16x16x32_f16(false, a, false, b, (short)0, acc0, false, false);
      }
      {
        int ko = hpad(kk + 32);
        v8h alo = *(const v8h*)&arow[ko + kh];
        v8h ahi = *(const v8h*)&arow[ko + kh + 16];
        v16h a = __builtin_shufflevector(alo, ahi, 0,1,2,3,4,5,6,7,8,9,10,11,12,13,14,15);
        v8h blo = *(const v8h*)&brow[kk + 32 + kh];
        v8h bhi = *(const v8h*)&brow[kk + 32 + kh + 16];
        v16h b = __builtin_shufflevector(blo, bhi, 0,1,2,3,4,5,6,7,8,9,10,11,12,13,14,15);
        acc1 = __builtin_amdgcn_wmma_f32_16x16x32_f16(false, a, false, b, (short)0, acc1, false, false);
      }
    }
    v8f acc = acc0 + acc1;

    // park gate tiles in LDS so one pass can combine r/z/n
    {
      int mrow = mi * 16 + ((lane >> 4) << 3);
      #pragma unroll
      for (int r = 0; r < 8; ++r)
        ghb[(g * 32 + mrow + r) * 16 + (lane & 15)] = acc[r];
    }
    __syncthreads();

    // gates + state update for this WG's 32x16 slice
    for (int e = tid; e < 32 * 16; e += RT) {
      int m = e >> 4, jj = e & 15;
      int j = j0 + jj;
      long gxo = ((long)m * S_ + s) * (long)G3H;
      float xr = gx[gxo + j];
      float xz = gx[gxo + H_ + j];
      float xn = gx[gxo + 2 * H_ + j];
      float hr = ghb[(0 * 32 + m) * 16 + jj] + bhh[j];
      float hz = ghb[(1 * 32 + m) * 16 + jj] + bhh[H_ + j];
      float hn = ghb[(2 * 32 + m) * 16 + jj] + bhh[2 * H_ + j];
      float r = 1.f / (1.f + __expf(-(xr + hr)));
      float z = 1.f / (1.f + __expf(-(xz + hz)));
      float nn = tanhf(xn + r * hn);
      float hp = (float)hb[m * LDH + hpad(j)];
      float hv = (1.f - z) * nn + z * hp;
      hnxt[m * H_ + j] = (_Float16)hv;
      if (seq)  seq[((long)m * S_ + s) * H2 + dir * H_ + j] = (_Float16)hv;
      if (outp && step == S_ - 1) outp[m * H2 + dir * H_ + j] = hv;
    }
    __syncthreads();

    // per-direction grid barrier (monotonic counter, zeroed before launch)
    if (tid == 0) {
      __threadfence();
      __hip_atomic_fetch_add(ctr, 1u, __ATOMIC_RELEASE, __HIP_MEMORY_SCOPE_AGENT);
      unsigned target = (unsigned)NWG * (unsigned)(step + 1);
      while (__hip_atomic_load(ctr, __ATOMIC_ACQUIRE, __HIP_MEMORY_SCOPE_AGENT) < target)
        __builtin_amdgcn_s_sleep(1);
    }
    __syncthreads();
  }
}

// ---------------- host ----------------
extern "C" void kernel_launch(void* const* d_in, const int* in_sizes, int n_in,
                              void* d_out, int out_size, void* d_ws, size_t ws_size,
                              hipStream_t stream) {
  (void)in_sizes; (void)n_in; (void)out_size; (void)ws_size;

  const float* x      = (const float*)d_in[0];
  const float* wih_f0 = (const float*)d_in[1];
  const float* whh_f0 = (const float*)d_in[2];
  const float* bih_f0 = (const float*)d_in[3];
  const float* bhh_f0 = (const float*)d_in[4];
  const float* wih_b0 = (const float*)d_in[5];
  const float* whh_b0 = (const float*)d_in[6];
  const float* bih_b0 = (const float*)d_in[7];
  const float* bhh_b0 = (const float*)d_in[8];
  const float* wih_f1 = (const float*)d_in[9];
  const float* whh_f1 = (const float*)d_in[10];
  const float* bih_f1 = (const float*)d_in[11];
  const float* bhh_f1 = (const float*)d_in[12];
  const float* wih_b1 = (const float*)d_in[13];
  const float* whh_b1 = (const float*)d_in[14];
  const float* bih_b1 = (const float*)d_in[15];
  const float* bhh_b1 = (const float*)d_in[16];

  char* ws = (char*)d_ws;
  size_t off = 0;
  auto alloc = [&](size_t bytes) -> void* {
    void* p = ws + off;
    off += (bytes + 255) & ~(size_t)255;
    return p;
  };

  _Float16* xh     = (_Float16*)alloc((size_t)MTOT * I_ * 2);
  _Float16* wihF0h = (_Float16*)alloc((size_t)G3H * I_ * 2);
  _Float16* wihB0h = (_Float16*)alloc((size_t)G3H * I_ * 2);
  _Float16* whhF0h = (_Float16*)alloc((size_t)G3H * H_ * 2);
  _Float16* whhB0h = (_Float16*)alloc((size_t)G3H * H_ * 2);
  _Float16* wihF1h = (_Float16*)alloc((size_t)G3H * H2 * 2);
  _Float16* wihB1h = (_Float16*)alloc((size_t)G3H * H2 * 2);
  _Float16* whhF1h = (_Float16*)alloc((size_t)G3H * H_ * 2);
  _Float16* whhB1h = (_Float16*)alloc((size_t)G3H * H_ * 2);
  float*    gxF    = (float*)alloc((size_t)MTOT * G3H * 4);
  float*    gxB    = (float*)alloc((size_t)MTOT * G3H * 4);
  _Float16* seq0   = (_Float16*)alloc((size_t)MTOT * H2 * 2);
  _Float16* hbufs  = (_Float16*)alloc((size_t)2 * 2 * B_ * H_ * 2);
  unsigned* ctrs   = (unsigned*)alloc(256);

  cvt_f32_f16<<<1024, 256, 0, stream>>>(x, xh, (long)MTOT * I_);
  cvt_f32_f16<<<512, 256, 0, stream>>>(wih_f0, wihF0h, (long)G3H * I_);
  cvt_f32_f16<<<512, 256, 0, stream>>>(wih_b0, wihB0h, (long)G3H * I_);
  cvt_f32_f16<<<512, 256, 0, stream>>>(whh_f0, whhF0h, (long)G3H * H_);
  cvt_f32_f16<<<512, 256, 0, stream>>>(whh_b0, whhB0h, (long)G3H * H_);
  cvt_f32_f16<<<512, 256, 0, stream>>>(wih_f1, wihF1h, (long)G3H * H2);
  cvt_f32_f16<<<512, 256, 0, stream>>>(wih_b1, wihB1h, (long)G3H * H2);
  cvt_f32_f16<<<512, 256, 0, stream>>>(whh_f1, whhF1h, (long)G3H * H_);
  cvt_f32_f16<<<512, 256, 0, stream>>>(whh_b1, whhB1h, (long)G3H * H_);

  size_t smem = (size_t)(48 + 32) * LDH * 2 + 3 * 32 * 16 * 4;
  (void)hipFuncSetAttribute(reinterpret_cast<const void*>(bigru_recur),
                            hipFuncAttributeMaxDynamicSharedMemorySize, (int)smem);

  dim3 gg(G3H / BN, MTOT / BM, 2);

  // ---- layer 0 ----
  gemm_gx<<<gg, 256, 0, stream>>>(xh, wihF0h, wihB0h, bih_f0, bih_b0, gxF, gxB, I_);
  (void)hipMemsetAsync(hbufs, 0, (size_t)2 * 2 * B_ * H_ * 2, stream);
  (void)hipMemsetAsync(ctrs, 0, 256, stream);
  bigru_recur<<<2 * NWG, RT, smem, stream>>>(gxF, gxB, whhF0h, whhB0h, bhh_f0, bhh_b0,
                                             seq0, nullptr, hbufs, ctrs);

  // ---- layer 1 (gx buffers reused; input is layer-0 sequence, K = 2H) ----
  gemm_gx<<<gg, 256, 0, stream>>>(seq0, wihF1h, wihB1h, bih_f1, bih_b1, gxF, gxB, H2);
  (void)hipMemsetAsync(hbufs, 0, (size_t)2 * 2 * B_ * H_ * 2, stream);
  (void)hipMemsetAsync(ctrs, 0, 256, stream);
  bigru_recur<<<2 * NWG, RT, smem, stream>>>(gxF, gxB, whhF1h, whhB1h, bhh_f1, bhh_b1,
                                             nullptr, (float*)d_out, hbufs, ctrs);
}
